// MultiHeadContrastive_12429635354742
// MI455X (gfx1250) — matrix operand order repaired and checked by reading, not compiled, not run on hardware
//
#include <hip/hip_runtime.h>
#include <hip/hip_bf16.h>
#include <math.h>
#include <stdint.h>

typedef __attribute__((ext_vector_type(16))) __bf16 v16bf;
typedef __attribute__((ext_vector_type(8)))  __bf16 v8bf;
typedef __attribute__((ext_vector_type(8)))  float  v8f;
typedef __attribute__((ext_vector_type(4))) unsigned int u32x4;
typedef __attribute__((ext_vector_type(8))) int i32x8;
typedef __attribute__((ext_vector_type(4))) int i32x4;

namespace {

constexpr int   N_ROI = 8192;
constexpr int   C_IN  = 1024;
constexpr int   H_MID = 256;
constexpr int   DD_FG = 64;
constexpr int   DD_CL = 128;
constexpr float EPSF  = 1e-8f;
constexpr float INV_TAU = 5.0f;   // 1 / 0.2
constexpr float NEGB  = -1e9f;

union FragA { v16bf v; v8bf h[2]; };

// ---------------- CDNA5 async helpers ----------------

// Per-lane async global->LDS 16B copy (ASYNCcnt).  GVS mode: saddr base + 32-bit vgpr offset.
__device__ __forceinline__ void async_copy_b128(unsigned lds_addr, const void* sbase, unsigned byte_off) {
    asm volatile("global_load_async_to_lds_b128 %0, %1, %2 offset:0"
                 :: "v"(lds_addr), "v"(byte_off), "s"(sbase)
                 : "memory");
}

// Tensor Data Mover: 2D tile (tile_x elements of 2 bytes, tile_y rows, row stride in elements).
// D# groups per CDNA5 ISA 8.3/8.4; groups 2/3 zero (2D, tile_dim2 = 0).
// amdgpu-toolchain (clang-23) 6-arg builtin form.
__device__ __forceinline__ void tdm_load_2d(unsigned lds_addr, const void* gptr,
                                            unsigned tile_x, unsigned tile_y,
                                            unsigned long long stride_elems) {
    unsigned long long ga = (unsigned long long)(uintptr_t)gptr;
    u32x4 g0;
    g0[0] = 1u;                                                   // count=1, user mode
    g0[1] = lds_addr;                                             // lds_addr [63:32]
    g0[2] = (unsigned)ga;                                         // global_addr [95:64]
    g0[3] = (unsigned)((ga >> 32) & 0x01FFFFFFull) | (2u << 30);  // addr[56:32] | type=2
    const unsigned td = 0x7FFFFFFFu;                              // huge tensor dims: no OOB clip
    i32x8 g1;
    g1[0] = (int)(1u << 16);                                      // data_size = 1 (2 bytes)
    g1[1] = (int)((td & 0xFFFFu) << 16);                          // tensor_dim0[15:0] @ bits 63:48
    g1[2] = (int)(((td >> 16) & 0xFFFFu) | ((td & 0xFFFFu) << 16));        // td0 hi | td1 lo
    g1[3] = (int)(((td >> 16) & 0xFFFFu) | ((tile_x & 0xFFFFu) << 16));    // td1 hi | tile_dim0
    g1[4] = (int)(tile_y & 0xFFFFu);                              // tile_dim1 | tile_dim2=0
    g1[5] = (int)(unsigned)stride_elems;                          // tensor_dim0_stride[31:0]
    g1[6] = (int)((stride_elems >> 32) & 0xFFFFull);              // stride hi | dim1_stride lo = 0
    g1[7] = 0;
    i32x4 z4 = {};
    i32x8 z8 = {};
    __builtin_amdgcn_tensor_load_to_lds(g0, g1, z4, z4, z8, 0);
}

__device__ __forceinline__ unsigned lds_raw(const void* p) {
    return (unsigned)(uintptr_t)p;   // low 32 bits of generic pointer = LDS byte address
}

// ---------------- conversion kernels ----------------

__global__ void cvt_f32_bf16(const float* __restrict__ in, __bf16* __restrict__ out, int n) {
    int i = blockIdx.x * blockDim.x + threadIdx.x;
    if (i < n) out[i] = (__bf16)in[i];
}

// in: [K, Nn] row-major f32  ->  out: [Nn, K] row-major bf16
__global__ void transpose_cvt(const float* __restrict__ in, __bf16* __restrict__ out, int K, int Nn) {
    int i = blockIdx.x * blockDim.x + threadIdx.x;
    if (i < K * Nn) {
        int k = i / Nn;
        int n = i - k * Nn;
        out[(size_t)n * K + k] = (__bf16)in[i];
    }
}

// ---------------- GEMM1: LDS-staged, double-buffered, async + TDM ----------------
// Out[8192,256] = relu(A[8192,1024] @ Bt[256,1024]^T + bias), bf16 out.
// Block: 256 thr (8 waves) -> 128x64 tile.  A staged via global_load_async_to_lds_b128,
// B staged via TDM tensor_load_to_lds (wave 0).  2 buffers x (16KB A + 8KB B).

__global__ __launch_bounds__(256)
void wmma_gemm1_async(const __bf16* __restrict__ A, const __bf16* __restrict__ Bt,
                      const float* __restrict__ bias, __bf16* __restrict__ outB) {
    constexpr int K = C_IN;          // 1024
    constexpr int NOUT = H_MID;      // 256
    constexpr int KC = 64;
    constexpr int STAGES = K / KC;   // 16

    __shared__ __align__(16) __bf16 tileA[2][128 * KC];
    __shared__ __align__(16) __bf16 tileB[2][64 * KC];

    const int tid  = threadIdx.x;
    const int lane = tid & 31;
    const int wave = tid >> 5;
    const int l15  = lane & 15;
    const int half = lane >> 4;
    const int mb   = blockIdx.x >> 2;     // 64 row blocks of 128
    const int ng   = blockIdx.x & 3;      // 4 col groups of 64
    const int rowBase = mb * 128;

    auto stageIssue = [&](int s, int buf) {
        const unsigned k0 = (unsigned)(s * KC);
        // A tile: 128 rows x 64 cols bf16 = 1024 x 16B segments, 4 per thread
        #pragma unroll
        for (int i = 0; i < 4; ++i) {
            int g   = tid + i * 256;
            int row = g >> 3;
            int c16 = g & 7;
            unsigned goff  = (unsigned)(((unsigned)(rowBase + row) * K + k0) * 2u + (unsigned)c16 * 16u);
            unsigned laddr = lds_raw(&tileA[buf][row * KC + c16 * 8]);
            async_copy_b128(laddr, A, goff);
        }
        // B tile: 64 rows (out cols) x 64 K-elements, row stride K -- one TDM descriptor
        if (wave == 0) {
            tdm_load_2d(lds_raw(&tileB[buf][0]),
                        Bt + (size_t)(ng * 64) * K + k0,
                        KC, 64, (unsigned long long)K);
        }
    };

    stageIssue(0, 0);
    v8f acc[4] = {};
    for (int s = 0; s < STAGES; ++s) {
        const int buf = s & 1;
        if (s + 1 < STAGES) {
            stageIssue(s + 1, buf ^ 1);
            asm volatile("s_wait_asynccnt 0x4" ::: "memory");     // own 4 copies of stage s done
            if (wave == 0) __builtin_amdgcn_s_wait_tensorcnt(1);  // stage s TDM done
        } else {
            asm volatile("s_wait_asynccnt 0x0" ::: "memory");
            if (wave == 0) __builtin_amdgcn_s_wait_tensorcnt(0);
        }
        __syncthreads();

        const __bf16* arow_l = &tileA[buf][(wave * 16 + l15) * KC];
        #pragma unroll
        for (int kk = 0; kk < KC; kk += 32) {
            FragA a;
            a.h[0] = *(const v8bf*)(arow_l + kk + half * 16);
            a.h[1] = *(const v8bf*)(arow_l + kk + half * 16 + 8);
            #pragma unroll
            for (int t = 0; t < 4; ++t) {
                v16bf b = *(const v16bf*)(&tileB[buf][(t * 16 + l15) * KC + kk + half * 16]);
                acc[t] = __builtin_amdgcn_wmma_f32_16x16x32_bf16(false, a.v, false, b, (short)0, acc[t], false, false);
            }
        }
        __syncthreads();   // safe to overwrite buf next iteration
    }

    const int mbase = rowBase + wave * 16 + 8 * half;
    #pragma unroll
    for (int t = 0; t < 4; ++t) {
        int n = ng * 64 + t * 16 + l15;
        float bv = bias[n];
        #pragma unroll
        for (int r = 0; r < 8; ++r) {
            float val = fmaxf(acc[t][r] + bv, 0.0f);
            outB[(size_t)(mbase + r) * NOUT + n] = (__bf16)val;
        }
    }
}

// ---------------- GEMM2 (small K): direct-global WMMA ----------------

template<int KDIM>
__global__ __launch_bounds__(256)
void wmma_gemm_direct(const __bf16* __restrict__ A, const __bf16* __restrict__ Bt,
                      const float* __restrict__ bias, float* __restrict__ outF, int Nout) {
    const int lane = threadIdx.x & 31;
    const int wave = threadIdx.x >> 5;
    const int l15  = lane & 15;
    const int half = lane >> 4;
    const int ngroups = Nout >> 6;
    const int tile  = blockIdx.x * 8 + wave;
    const int mtile = tile / ngroups;
    const int ng    = tile - mtile * ngroups;

    const __bf16* __restrict__ arow  = A  + (size_t)(mtile * 16 + l15) * KDIM;
    const __bf16* __restrict__ bbase = Bt + (size_t)(ng * 64 + l15) * KDIM + half * 16;

    v8f acc[4] = {};
    for (int k0 = 0; k0 < KDIM; k0 += 32) {
        FragA a;
        a.h[0] = *(const v8bf*)(arow + k0 + half * 16);
        a.h[1] = *(const v8bf*)(arow + k0 + half * 16 + 8);
        v16bf b0 = *(const v16bf*)(bbase + k0);
        v16bf b1 = *(const v16bf*)(bbase + k0 + 16 * KDIM);
        v16bf b2 = *(const v16bf*)(bbase + k0 + 32 * KDIM);
        v16bf b3 = *(const v16bf*)(bbase + k0 + 48 * KDIM);
        acc[0] = __builtin_amdgcn_wmma_f32_16x16x32_bf16(false, a.v, false, b0, (short)0, acc[0], false, false);
        acc[1] = __builtin_amdgcn_wmma_f32_16x16x32_bf16(false, a.v, false, b1, (short)0, acc[1], false, false);
        acc[2] = __builtin_amdgcn_wmma_f32_16x16x32_bf16(false, a.v, false, b2, (short)0, acc[2], false, false);
        acc[3] = __builtin_amdgcn_wmma_f32_16x16x32_bf16(false, a.v, false, b3, (short)0, acc[3], false, false);
    }
    const int mbase = mtile * 16 + 8 * half;
    #pragma unroll
    for (int t = 0; t < 4; ++t) {
        int n = ng * 64 + t * 16 + l15;
        float bv = bias[n];
        #pragma unroll
        for (int r = 0; r < 8; ++r)
            outF[(size_t)(mbase + r) * Nout + n] = acc[t][r] + bv;
    }
}

// ---------------- row normalize f32 -> bf16 ----------------

template<int D>
__global__ __launch_bounds__(256)
void normalize_rows(const float* __restrict__ Zf, __bf16* __restrict__ Zb) {
    const int lane = threadIdx.x & 31;
    const int wave = threadIdx.x >> 5;
    const int row  = blockIdx.x * 8 + wave;
    const float* zr = Zf + (size_t)row * D;
    float ss = 0.f;
    #pragma unroll
    for (int k = lane; k < D; k += 32) { float v = zr[k]; ss += v * v; }
    #pragma unroll
    for (int mask = 16; mask > 0; mask >>= 1) ss += __shfl_xor(ss, mask, 32);
    float inv = 1.0f / fmaxf(sqrtf(ss), EPSF);
    #pragma unroll
    for (int k = lane; k < D; k += 32) Zb[(size_t)row * D + k] = (__bf16)(zr[k] * inv);
}

// ---------------- fused similarity, TDM-staged j-tiles ----------------
// 8 waves/block share each staged chunk of 64 j-rows (one TDM descriptor, double buffered).
//   fg head (CLS=false):  o0 = denom = sum_{j!=i} exp(sim),  o1 = numer over fg cols
//   cls head (CLS=true):  o0 = S;  o1 = P = sum posc*(diag?NEG:sim);  o2 = n_posc

template<int D, bool CLS>
__global__ __launch_bounds__(256)
void wmma_sim_tdm(const __bf16* __restrict__ Z, const int* __restrict__ labels,
                  float* __restrict__ o0, float* __restrict__ o1, float* __restrict__ o2, int N) {
    constexpr int KF = D / 32;
    constexpr int JC = 64;                        // j-rows per stage
    __shared__ __align__(16) __bf16 tileZ[2][JC * D];

    const int lane = threadIdx.x & 31;
    const int wave = threadIdx.x >> 5;
    const int l15  = lane & 15;
    const int half = lane >> 4;
    const int itile = blockIdx.x * 8 + wave;

    // A fragments pinned in registers for the entire j sweep
    FragA af[KF];
    const __bf16* __restrict__ arow = Z + (size_t)(itile * 16 + l15) * D;
    #pragma unroll
    for (int kf = 0; kf < KF; ++kf) {
        af[kf].h[0] = *(const v8bf*)(arow + kf * 32 + half * 16);
        af[kf].h[1] = *(const v8bf*)(arow + kf * 32 + half * 16 + 8);
    }

    int li[8];
    if (CLS) {
        #pragma unroll
        for (int r = 0; r < 8; ++r) li[r] = labels[itile * 16 + 8 * half + r];
    }

    auto issue = [&](int st, int buf) {
        if (wave == 0)
            tdm_load_2d(lds_raw(&tileZ[buf][0]), Z + (size_t)st * JC * D,
                        D, JC, (unsigned long long)D);
    };

    float s0[8] = {}, s1[8] = {}, s2[8] = {};
    const int STAGES = N / JC;     // 128
    issue(0, 0);
    for (int st = 0; st < STAGES; ++st) {
        const int buf = st & 1;
        if (st + 1 < STAGES) {
            issue(st + 1, buf ^ 1);
            if (wave == 0) __builtin_amdgcn_s_wait_tensorcnt(1);
        } else {
            if (wave == 0) __builtin_amdgcn_s_wait_tensorcnt(0);
        }
        __syncthreads();

        #pragma unroll
        for (int jl = 0; jl < JC / 16; ++jl) {
            const int jt = st * (JC / 16) + jl;
            const __bf16* brow = &tileZ[buf][(jl * 16 + l15) * D + half * 16];
            v8f c = {};
            #pragma unroll
            for (int kf = 0; kf < KF; ++kf) {
                v16bf b = *(const v16bf*)(brow + kf * 32);
                c = __builtin_amdgcn_wmma_f32_16x16x32_bf16(false, af[kf].v, false, b, (short)0, c, false, false);
            }
            const int j  = jt * 16 + l15;
            const int lj = labels[j];
            #pragma unroll
            for (int r = 0; r < 8; ++r) {
                const int m = itile * 16 + 8 * half + r;
                float sim = c[r] * INV_TAU;
                float e   = __expf(sim);
                bool diag = (m == j);
                if (!CLS) {
                    s0[r] += diag ? 0.f : e;
                    s1[r] += (!diag && lj > 0) ? e : 0.f;
                } else {
                    s0[r] += diag ? 0.f : e;
                    bool posc = (lj == li[r]) && (lj != -1) && (lj != 0);
                    s1[r] += posc ? (diag ? NEGB : sim) : 0.f;
                    s2[r] += posc ? 1.f : 0.f;
                }
            }
        }
        __syncthreads();
    }

    #pragma unroll
    for (int r = 0; r < 8; ++r) {
        #pragma unroll
        for (int mask = 1; mask <= 8; mask <<= 1) {
            s0[r] += __shfl_xor(s0[r], mask, 32);
            s1[r] += __shfl_xor(s1[r], mask, 32);
            if (CLS) s2[r] += __shfl_xor(s2[r], mask, 32);
        }
    }
    if (l15 == 0) {
        #pragma unroll
        for (int r = 0; r < 8; ++r) {
            int m = itile * 16 + 8 * half + r;
            o0[m] = s0[r];
            o1[m] = s1[r];
            if (CLS) o2[m] = s2[r];
        }
    }
}

// ---------------- final weighted reduction (single block) ----------------

__global__ __launch_bounds__(256)
void loss_kernel(const float* __restrict__ fden, const float* __restrict__ fnum,
                 const float* __restrict__ cS, const float* __restrict__ cP,
                 const float* __restrict__ cNp,
                 const int* __restrict__ labels, const float* __restrict__ ious,
                 float* __restrict__ out, int N) {
    __shared__ float red[256];
    const int t = threadIdx.x;

    float cnt = 0.f;
    for (int i = t; i < N; i += 256) cnt += (labels[i] > 0) ? 1.f : 0.f;
    red[t] = cnt; __syncthreads();
    for (int s = 128; s > 0; s >>= 1) { if (t < s) red[t] += red[t + s]; __syncthreads(); }
    const float fgtot = red[0];
    __syncthreads();

    float sw0 = 0.f, swl0 = 0.f, sw1 = 0.f, swl1 = 0.f;
    for (int i = t; i < N; i += 256) {
        const int lab = labels[i];
        const bool fg = lab > 0;
        const float iou = ious[i];
        const float iw  = (iou > 0.5f) ? iou : 0.f;

        float npos = fgtot - (fg ? 1.f : 0.f);
        float li = -logf((fnum[i] + EPSF) / (fden[i] + 2.f * EPSF));
        float w  = (fg && npos > 0.f) ? iw : 0.f;
        sw0 += w; swl0 += w * li;

        float np = cNp[i];
        bool posDiag = (lab != -1) && (lab != 0);
        float npOff = np - (posDiag ? 1.f : 0.f);
        float logden = logf(cS[i]);
        float Lzi = -(cP[i] - npOff * logden) / (np + EPSF);
        float wc  = (fg && np > 0.f) ? iw : 0.f;
        sw1 += wc; swl1 += wc * Lzi;
    }

    red[t] = sw0;  __syncthreads();
    for (int s = 128; s > 0; s >>= 1) { if (t < s) red[t] += red[t + s]; __syncthreads(); }
    float rsw0 = red[0]; __syncthreads();
    red[t] = swl0; __syncthreads();
    for (int s = 128; s > 0; s >>= 1) { if (t < s) red[t] += red[t + s]; __syncthreads(); }
    float rswl0 = red[0]; __syncthreads();
    red[t] = sw1;  __syncthreads();
    for (int s = 128; s > 0; s >>= 1) { if (t < s) red[t] += red[t + s]; __syncthreads(); }
    float rsw1 = red[0]; __syncthreads();
    red[t] = swl1; __syncthreads();
    for (int s = 128; s > 0; s >>= 1) { if (t < s) red[t] += red[t + s]; __syncthreads(); }
    float rswl1 = red[0]; __syncthreads();

    if (t == 0) {
        out[0] = rswl0 / (rsw0 + EPSF);
        out[1] = rswl1 / (rsw1 + EPSF);
    }
}

} // namespace

extern "C" void kernel_launch(void* const* d_in, const int* in_sizes, int n_in,
                              void* d_out, int out_size, void* d_ws, size_t ws_size,
                              hipStream_t stream) {
    const float* roi     = (const float*)d_in[0];
    const int*   labels  = (const int*)d_in[1];
    const float* ious    = (const float*)d_in[2];
    const float* fg_w1   = (const float*)d_in[3];
    const float* fg_b1   = (const float*)d_in[4];
    const float* fg_w2   = (const float*)d_in[5];
    const float* fg_b2   = (const float*)d_in[6];
    const float* cls_w1  = (const float*)d_in[7];
    const float* cls_b1  = (const float*)d_in[8];
    const float* cls_w2  = (const float*)d_in[9];
    const float* cls_b2  = (const float*)d_in[10];
    float* out = (float*)d_out;

    char* ws = (char*)d_ws;
    size_t off = 0;
    auto alloc = [&](size_t bytes) -> void* {
        off = (off + 255) & ~(size_t)255;
        void* p = ws + off;
        off += bytes;
        return p;
    };

    __bf16* Abf    = (__bf16*)alloc((size_t)N_ROI * C_IN * 2);
    __bf16* W1tFg  = (__bf16*)alloc((size_t)H_MID * C_IN * 2);
    __bf16* W1tCl  = (__bf16*)alloc((size_t)H_MID * C_IN * 2);
    __bf16* W2tFg  = (__bf16*)alloc((size_t)DD_FG * H_MID * 2);
    __bf16* W2tCl  = (__bf16*)alloc((size_t)DD_CL * H_MID * 2);
    __bf16* Hfg    = (__bf16*)alloc((size_t)N_ROI * H_MID * 2);
    __bf16* Hcl    = (__bf16*)alloc((size_t)N_ROI * H_MID * 2);
    float*  ZfFg   = (float*)alloc((size_t)N_ROI * DD_FG * 4);
    float*  ZfCl   = (float*)alloc((size_t)N_ROI * DD_CL * 4);
    __bf16* ZbFg   = (__bf16*)alloc((size_t)N_ROI * DD_FG * 2);
    __bf16* ZbCl   = (__bf16*)alloc((size_t)N_ROI * DD_CL * 2);
    float*  fgDen  = (float*)alloc((size_t)N_ROI * 4);
    float*  fgNum  = (float*)alloc((size_t)N_ROI * 4);
    float*  clsS   = (float*)alloc((size_t)N_ROI * 4);
    float*  clsP   = (float*)alloc((size_t)N_ROI * 4);
    float*  clsNp  = (float*)alloc((size_t)N_ROI * 4);

    // 1) convert inputs to bf16 (weights pre-transposed to [Nout, K])
    cvt_f32_bf16<<<(N_ROI * C_IN) / 256, 256, 0, stream>>>(roi, Abf, N_ROI * C_IN);
    transpose_cvt<<<(C_IN * H_MID) / 256, 256, 0, stream>>>(fg_w1,  W1tFg, C_IN,  H_MID);
    transpose_cvt<<<(C_IN * H_MID) / 256, 256, 0, stream>>>(cls_w1, W1tCl, C_IN,  H_MID);
    transpose_cvt<<<(H_MID * DD_FG) / 256, 256, 0, stream>>>(fg_w2,  W2tFg, H_MID, DD_FG);
    transpose_cvt<<<(H_MID * DD_CL) / 256, 256, 0, stream>>>(cls_w2, W2tCl, H_MID, DD_CL);

    // 2) GEMM1 + relu (async/TDM staged): 256 blocks = 64 row-blocks x 4 col-groups
    wmma_gemm1_async<<<256, 256, 0, stream>>>(Abf, W1tFg, fg_b1,  Hfg);
    wmma_gemm1_async<<<256, 256, 0, stream>>>(Abf, W1tCl, cls_b1, Hcl);

    // 3) GEMM2: [8192,256]@[256,{64,128}] -> f32 pre-norm z
    wmma_gemm_direct<H_MID><<<((N_ROI / 16) * (DD_FG / 64)) / 8, 256, 0, stream>>>(Hfg, W2tFg, fg_b2,  ZfFg, DD_FG);
    wmma_gemm_direct<H_MID><<<((N_ROI / 16) * (DD_CL / 64)) / 8, 256, 0, stream>>>(Hcl, W2tCl, cls_b2, ZfCl, DD_CL);

    // 4) row-normalize -> bf16 unit vectors
    normalize_rows<DD_FG><<<N_ROI / 8, 256, 0, stream>>>(ZfFg, ZbFg);
    normalize_rows<DD_CL><<<N_ROI / 8, 256, 0, stream>>>(ZfCl, ZbCl);

    // 5) fused z@z.T similarity + row reductions (TDM-staged j-chunks)
    wmma_sim_tdm<DD_FG, false><<<(N_ROI / 16) / 8, 256, 0, stream>>>(ZbFg, labels, fgDen, fgNum, nullptr, N_ROI);
    wmma_sim_tdm<DD_CL, true ><<<(N_ROI / 16) / 8, 256, 0, stream>>>(ZbCl, labels, clsS,  clsP,  clsNp,  N_ROI);

    // 6) final weighted scalar losses
    loss_kernel<<<1, 256, 0, stream>>>(fgDen, fgNum, clsS, clsP, clsNp, labels, ious, out, N_ROI);
    (void)in_sizes; (void)n_in; (void)out_size; (void)ws_size;
}
